// GAT_17179869184528
// MI455X (gfx1250) — matrix operand order repaired and checked by reading, not compiled, not run on hardware
//
#include <hip/hip_runtime.h>
#include <hip/hip_bf16.h>
#include <cstdint>

typedef __attribute__((ext_vector_type(16))) _Float16 v16h;
typedef __attribute__((ext_vector_type(8)))  _Float16 v8h;
typedef __attribute__((ext_vector_type(8)))  float    v8f;
typedef __attribute__((ext_vector_type(2)))  float    v2f;

#define SEQ    2048
#define FIN    512
#define FOUT   512
#define NB     8
#define ALPHA  0.2f
#define NEGINF -9e15f

// ---------------------------------------------------------------------------
// Kernel 1: Wh = X @ W  (full f32 precision via V_WMMA_F32_16X16X4_F32)
// block = 256 thr = 8 waves; block covers 32 rows x 512 cols; wave owns 64 cols
// ---------------------------------------------------------------------------
__global__ __launch_bounds__(256) void k_gemm1(const float* __restrict__ X,
                                               const float* __restrict__ W,
                                               float* __restrict__ Wh) {
  const int b    = blockIdx.y;
  const int s0   = blockIdx.x * 32;
  const int wave = threadIdx.x >> 5;
  const int lane = threadIdx.x & 31;
  const int l16  = lane & 15;
  const int hf   = lane >> 4;          // lane half selects K pair
  const int o0   = wave * 64;

  const float* Xrow0 = X + ((size_t)b * SEQ + s0 + l16) * FIN;
  const float* Xrow1 = Xrow0 + 16 * FIN;

  v8f acc[2][4] = {};

  for (int k0 = 0; k0 < FIN; k0 += 4) {
    // A frag (16x4 f32): lanes 0-15 K0,K1 ; lanes 16-31 K2,K3
    v2f a0 = *(const v2f*)(Xrow0 + k0 + hf * 2);
    v2f a1 = *(const v2f*)(Xrow1 + k0 + hf * 2);
#pragma unroll
    for (int nt = 0; nt < 4; ++nt) {
      const int oc = o0 + nt * 16 + l16;
      // B frag (4x16 f32): lane = col, VGPR0/1 = consecutive K of lane half
      v2f bf;
      bf.x = W[(size_t)(k0 + hf * 2)     * FOUT + oc];
      bf.y = W[(size_t)(k0 + hf * 2 + 1) * FOUT + oc];
      acc[0][nt] = __builtin_amdgcn_wmma_f32_16x16x4_f32(
          false, a0, false, bf, (short)0, acc[0][nt], false, false);
      acc[1][nt] = __builtin_amdgcn_wmma_f32_16x16x4_f32(
          false, a1, false, bf, (short)0, acc[1][nt], false, false);
    }
  }
#pragma unroll
  for (int mt = 0; mt < 2; ++mt)
#pragma unroll
    for (int nt = 0; nt < 4; ++nt)
#pragma unroll
      for (int r = 0; r < 8; ++r) {
        const int row = s0 + mt * 16 + hf * 8 + r;   // C layout: VGPR r -> M=r / M=8+r
        const int col = o0 + nt * 16 + l16;
        Wh[((size_t)b * SEQ + row) * FOUT + col] = acc[mt][nt][r];
      }
}

// ---------------------------------------------------------------------------
// Kernel 2: wh1 = Wh . a[:F], wh2 = Wh . a[F:]  (one wave per row)
// ---------------------------------------------------------------------------
__global__ __launch_bounds__(256) void k_rowdots(const float* __restrict__ Wh,
                                                 const float* __restrict__ a,
                                                 float* __restrict__ wh1,
                                                 float* __restrict__ wh2) {
  const int warp = threadIdx.x >> 5, lane = threadIdx.x & 31;
  const int row  = blockIdx.x * 8 + warp;             // [0, NB*SEQ)
  const float* p = Wh + (size_t)row * FOUT;
  float s1 = 0.f, s2 = 0.f;
  for (int i = lane; i < FOUT; i += 32) {
    const float v = p[i];
    s1 += v * a[i];
    s2 += v * a[FOUT + i];
  }
#pragma unroll
  for (int off = 16; off; off >>= 1) {
    s1 += __shfl_xor(s1, off, 32);
    s2 += __shfl_xor(s2, off, 32);
  }
  if (lane == 0) { wh1[row] = s1; wh2[row] = s2; }
}

// ---------------------------------------------------------------------------
// Kernel 3: WhT[b][o][t] = (f16) Wh[b][t][o]   (32x32 LDS tile transpose)
// ---------------------------------------------------------------------------
__global__ __launch_bounds__(256) void k_transpose(const float* __restrict__ Wh,
                                                   _Float16* __restrict__ WhT) {
  __shared__ float tile[32][33];
  const int b = blockIdx.z;
  const int t0 = blockIdx.x * 32;   // SEQ dim
  const int o0 = blockIdx.y * 32;   // FOUT dim
  const int col = threadIdx.x & 31, rg = threadIdx.x >> 5;
#pragma unroll
  for (int i = 0; i < 4; ++i) {
    const int r = rg + i * 8;
    tile[r][col] = Wh[((size_t)b * SEQ + t0 + r) * FOUT + o0 + col];
  }
  __syncthreads();
#pragma unroll
  for (int i = 0; i < 4; ++i) {
    const int r = rg + i * 8;       // r = o within tile
    WhT[((size_t)b * FOUT + o0 + r) * SEQ + t0 + col] = (_Float16)tile[col][r];
  }
}

// ---------------------------------------------------------------------------
// Kernel 4: per-column (axis=1) online softmax stats over masked leaky-relu
// ---------------------------------------------------------------------------
__global__ __launch_bounds__(256) void k_colstats(const float* __restrict__ adj,
                                                  const float* __restrict__ wh1,
                                                  const float* __restrict__ wh2,
                                                  float* __restrict__ colmax,
                                                  float* __restrict__ rcolsum) {
  const int b = blockIdx.y;
  const int t = blockIdx.x * 256 + threadIdx.x;
  const float* ap = adj + (size_t)b * SEQ * SEQ + t;
  const float* w1 = wh1 + b * SEQ;
  const float  w2t = wh2[b * SEQ + t];
  float m = -INFINITY, l = 0.f;
  for (int s = 0; s < SEQ; ++s) {
    const float av = ap[(size_t)s * SEQ];
    float e = w1[s] + w2t;
    e = (e > 0.f) ? e : ALPHA * e;
    const float v = (av > 0.5f) ? e : NEGINF;
    if (v > m) { l = l * __expf(m - v); m = v; }
    l += __expf(v - m);
  }
  colmax[b * SEQ + t]  = m;
  rcolsum[b * SEQ + t] = 1.0f / l;
}

// ---------------------------------------------------------------------------
// Kernel 5: out = elu( P @ Wh ), P rebuilt on the fly (adjacency read once)
// block = 256 thr = 8 waves; block covers 32 s-rows x all 512 o-cols.
// Per K-chunk(32): all threads build P(32x32) f16 in LDS; waves run f16 WMMA.
// ---------------------------------------------------------------------------
__global__ __launch_bounds__(256) void k_gemm2(const float* __restrict__ adj,
                                               const _Float16* __restrict__ WhT,
                                               const float* __restrict__ wh1,
                                               const float* __restrict__ wh2,
                                               const float* __restrict__ colmax,
                                               const float* __restrict__ rcolsum,
                                               float* __restrict__ out) {
  __shared__ __align__(16) _Float16 Pt[32 * 40];   // stride 40 halfs = 80B (16B aligned)
  const int b    = blockIdx.y;
  const int s0   = blockIdx.x * 32;
  const int tid  = threadIdx.x;
  const int wave = tid >> 5, lane = tid & 31, l16 = lane & 15, hf = lane >> 4;
  const int o0   = wave * 64;

  const int tc = tid & 31;          // t-within-chunk for P build
  const int sr = tid >> 5;          // s row group 0..7

  float w1v[4];
#pragma unroll
  for (int i = 0; i < 4; ++i) w1v[i] = wh1[b * SEQ + s0 + sr + i * 8];

  v8f acc[2][4] = {};
  const _Float16* Bbase = WhT + (size_t)b * FOUT * SEQ;

  for (int t0 = 0; t0 < SEQ; t0 += 32) {
    // ---- build attention tile P[32s x 32t] in LDS (f16) ----
    const int t   = t0 + tc;
    const float w2t = wh2[b * SEQ + t];
    const float cm  = colmax[b * SEQ + t];
    const float rs  = rcolsum[b * SEQ + t];
#pragma unroll
    for (int i = 0; i < 4; ++i) {
      const int s = sr + i * 8;
      const float av = adj[((size_t)b * SEQ + s0 + s) * SEQ + t];
      float e = w1v[i] + w2t;
      e = (e > 0.f) ? e : ALPHA * e;
      const float v = (av > 0.5f) ? e : NEGINF;
      Pt[s * 40 + tc] = (_Float16)(__expf(v - cm) * rs);
    }
    __syncthreads();

    // ---- A fragments from LDS (this code's k order, same as B: lane half = contiguous K16) ----
    v16h afrag[2];
#pragma unroll
    for (int mt = 0; mt < 2; ++mt) {
      const _Float16* ap = &Pt[(mt * 16 + l16) * 40 + hf * 16];
      const v8h lo = *(const v8h*)(ap);
      const v8h hi = *(const v8h*)(ap + 8);
#pragma unroll
      for (int i = 0; i < 8; ++i) { afrag[mt][i] = lo[i]; afrag[mt][i + 8] = hi[i]; }
    }
    // ---- B fragments from WhT (this code's k order, same as A: lane half = contiguous K16) ----
#pragma unroll
    for (int nt = 0; nt < 4; ++nt) {
      const _Float16* bp = Bbase + (size_t)(o0 + nt * 16 + l16) * SEQ + t0 + hf * 16;
      const v16h bfrag = *(const v16h*)(bp);
      acc[0][nt] = __builtin_amdgcn_wmma_f32_16x16x32_f16(
          false, afrag[0], false, bfrag, (short)0, acc[0][nt], false, false);
      acc[1][nt] = __builtin_amdgcn_wmma_f32_16x16x32_f16(
          false, afrag[1], false, bfrag, (short)0, acc[1][nt], false, false);
    }
    __syncthreads();
  }

  // ---- ELU epilogue + store ----
#pragma unroll
  for (int mt = 0; mt < 2; ++mt)
#pragma unroll
    for (int nt = 0; nt < 4; ++nt)
#pragma unroll
      for (int r = 0; r < 8; ++r) {
        float v = acc[mt][nt][r];
        v = (v > 0.f) ? v : (__expf(v) - 1.0f);
        const int row = s0 + mt * 16 + hf * 8 + r;
        const int col = o0 + nt * 16 + l16;
        out[((size_t)b * SEQ + row) * FOUT + col] = v;
      }
}

// ---------------------------------------------------------------------------
extern "C" void kernel_launch(void* const* d_in, const int* in_sizes, int n_in,
                              void* d_out, int out_size, void* d_ws, size_t ws_size,
                              hipStream_t stream) {
  const float* X   = (const float*)d_in[0];   // [8,2048,512]
  const float* adj = (const float*)d_in[1];   // [8,2048,2048]
  const float* W   = (const float*)d_in[2];   // [512,512]  (x @ W layout)
  const float* a   = (const float*)d_in[3];   // [1024,1]
  float* out = (float*)d_out;

  char* ws = (char*)d_ws;
  float*    Wh   = (float*)ws;     ws += (size_t)NB * SEQ * FOUT * sizeof(float);
  _Float16* WhT  = (_Float16*)ws;  ws += (size_t)NB * FOUT * SEQ * sizeof(_Float16);
  float* wh1     = (float*)ws;     ws += (size_t)NB * SEQ * sizeof(float);
  float* wh2     = (float*)ws;     ws += (size_t)NB * SEQ * sizeof(float);
  float* colmax  = (float*)ws;     ws += (size_t)NB * SEQ * sizeof(float);
  float* rcolsum = (float*)ws;     ws += (size_t)NB * SEQ * sizeof(float);

  k_gemm1    <<<dim3(SEQ / 32, NB),           256, 0, stream>>>(X, W, Wh);
  k_rowdots  <<<dim3(NB * SEQ / 8),           256, 0, stream>>>(Wh, a, wh1, wh2);
  k_transpose<<<dim3(SEQ / 32, FOUT / 32, NB),256, 0, stream>>>(Wh, WhT);
  k_colstats <<<dim3(SEQ / 256, NB),          256, 0, stream>>>(adj, wh1, wh2, colmax, rcolsum);
  k_gemm2    <<<dim3(SEQ / 32, NB),           256, 0, stream>>>(adj, WhT, wh1, wh2,
                                                                colmax, rcolsum, out);
}
